// Encoder_z0_ODE_RNN_41412074668412
// MI455X (gfx1250) — matrix operand order repaired and checked by reading, not compiled, not run on hardware
//
#include <hip/hip_runtime.h>
#include <hip/hip_bf16.h>

// ---------------------------------------------------------------------------
// ODE-RNN encoder (Latent-ODE Encoder_z0_ODE_RNN) for MI455X / gfx1250.
// B=512, T=128, D=256, L=512, H_ODE=512, Z0=512.
// Batch rows are independent through the recurrence -> each workgroup owns 16
// batch rows and runs the full 128-step scan internally; all GEMMs use
// v_wmma_f32_16x16x32_bf16 with fp32 accumulation. Register pressure is
// managed explicitly (launch_bounds + unroll caps, verified spill-free), and
// activations use branch-free v_exp/v_rcp sequences instead of libm tanhf.
// ---------------------------------------------------------------------------

typedef __bf16 bf16_t;
typedef bf16_t v16bf __attribute__((ext_vector_type(16)));
typedef float  v8f   __attribute__((ext_vector_type(8)));

#define T_STEPS 128
#define B_SZ    512
#define D_SZ    256
#define L_SZ    512
#define G_SZ    2048            // 4*L
#define PITCH   520             // LDS row pitch in bf16 elems (16B aligned)
#define THREADS 512             // 16 wave32
#define WAVES   16
#define TPW     2               // hidden-column 16-wide tiles per wave

union FragU { v16bf v; uint4 u[2]; };

__device__ __forceinline__ unsigned short f2bf(float f) {
  unsigned int u = __float_as_uint(f);
  u += 0x7FFFu + ((u >> 16) & 1u);          // round-to-nearest-even
  return (unsigned short)(u >> 16);
}
// branch-free tanh: (e^{2x}-1)/(e^{2x}+1), clamped so e^{2x} cannot overflow
__device__ __forceinline__ float tanh_fast(float x) {
  x = fminf(fmaxf(x, -15.0f), 15.0f);
  const float e = __expf(2.0f * x);
  return (e - 1.0f) * __builtin_amdgcn_rcpf(e + 1.0f);
}
// branch-free sigmoid
__device__ __forceinline__ float sigf(float x) {
  return __builtin_amdgcn_rcpf(1.0f + __expf(-x));
}

// A fragment: 16x32 bf16 tile from LDS (row-major, pitch PITCH).
// lanes 0-15: rows 0-15, K chunks {0..7,16..23}; lanes 16-31: K {8..15,24..31}.
__device__ __forceinline__ v16bf load_a_lds(const unsigned short* S, int lane, int k0) {
  FragU ua;
  const unsigned short* p = S + (lane & 15) * PITCH + k0 + ((lane >> 4) << 3);
  ua.u[0] = *(const uint4*)(p);
  ua.u[1] = *(const uint4*)(p + 16);
  return ua.v;
}

// B fragment: 32x16 bf16 tile, B[k,n] = W[nrow0+n, k] (W row-major [N,Kdim]).
// lane l: col = l&15, K chunks {0..7,16..23} (lanes 0-15) / {8..15,24..31} (lanes 16-31).
__device__ __forceinline__ v16bf load_b_g(const unsigned short* __restrict__ W,
                                          int Kdim, int nrow0, int lane, int k0) {
  FragU ub;
  const unsigned short* p = W + (size_t)(nrow0 + (lane & 15)) * Kdim + k0 + ((lane >> 4) << 3);
  ub.u[0] = *(const uint4*)(p);
  ub.u[1] = *(const uint4*)(p + 16);
  return ub.v;
}

#define WMMA_BF16(A, B, C) \
  __builtin_amdgcn_wmma_f32_16x16x32_bf16(false, (A), false, (B), (short)0, (C), false, false)

__global__ __launch_bounds__(THREADS, 1) void ode_rnn_scan_kernel(
    const float* __restrict__ truth, const float* __restrict__ maskp,
    const float* __restrict__ ts,
    const unsigned short* __restrict__ Wih, const unsigned short* __restrict__ Whh,
    const unsigned short* __restrict__ Wo1, const unsigned short* __restrict__ Wo2,
    const float* __restrict__ b_ih, const float* __restrict__ b_hh,
    const float* __restrict__ bo1, const float* __restrict__ bo2,
    float* __restrict__ xfin)
{
  __shared__ __align__(16) unsigned short xs[16 * PITCH];  // input tile (bf16)
  __shared__ __align__(16) unsigned short xa[16 * PITCH];  // x state staging (bf16)
  __shared__ __align__(16) unsigned short h1[16 * PITCH];  // ODE hidden (bf16)
  __shared__ float s_bo1[L_SZ], s_bo2[L_SZ], s_bsum[G_SZ], s_dt[16];
  __shared__ int   s_mflag[16];

  const int tid   = threadIdx.x;
  const int lane  = tid & 31;
  const int wv    = tid >> 5;
  const int half  = lane >> 4;
  const int lq    = lane & 15;
  const int bbase = blockIdx.x * 16;

  for (int i = tid; i < L_SZ; i += THREADS) { s_bo1[i] = bo1[i]; s_bo2[i] = bo2[i]; }
  for (int i = tid; i < G_SZ; i += THREADS) s_bsum[i] = b_ih[i] + b_hh[i];

  // persistent state in WMMA C-fragment layout: element r of tile jj is
  // (row = half*8 + r, col = (wv*TPW+jj)*16 + lq)
  v8f xst[TPW], cst[TPW];
  #pragma unroll
  for (int j = 0; j < TPW; j++) {
    #pragma unroll
    for (int r = 0; r < 8; r++) { xst[j][r] = 0.0f; cst[j][r] = 0.0f; }
  }
  __syncthreads();

  for (int t = 0; t < T_STEPS; t++) {
    const int tt = T_STEPS - 1 - t;   // backwards in time
    if (tid < 16) {
      const int b = bbase + tid;
      s_dt[tid]   = (t == 0) ? -0.01f
                             : (ts[(size_t)b * T_STEPS + tt] - ts[(size_t)b * T_STEPS + tt + 1]);
      s_mflag[tid] = 0;
    }
    // stage current (post-mask) x as bf16 A operand for the ODE GEMM
    #pragma unroll
    for (int jj = 0; jj < TPW; jj++) {
      const int n = (wv * TPW + jj) * 16 + lq;
      #pragma unroll
      for (int r = 0; r < 8; r++) xa[(half * 8 + r) * PITCH + n] = f2bf(xst[jj][r]);
    }
    __syncthreads();

    // load xi = concat(truth, mask) tile [16 x 512] -> bf16, set mask flags
    {
      const float* tr = truth + (size_t)bbase * T_STEPS * D_SZ + (size_t)tt * D_SZ;
      const float* mk = maskp + (size_t)bbase * T_STEPS * D_SZ + (size_t)tt * D_SZ;
      #pragma unroll 4
      for (int it = 0; it < (16 * 512) / THREADS; ++it) {
        const int idx = tid + it * THREADS;
        const int row = idx >> 9;
        const int col = idx & 511;
        float v;
        if (col < D_SZ) {
          v = tr[(size_t)row * T_STEPS * D_SZ + col];
        } else {
          v = mk[(size_t)row * T_STEPS * D_SZ + (col - D_SZ)];
          if (v > 0.0f) atomicOr(&s_mflag[row], 1);
        }
        xs[row * PITCH + col] = f2bf(v);
      }
    }
    __syncthreads();

    // ===== ODE stage 1: h1 = tanh(x @ Wo1^T + bo1)   [16,512]x[512,512]
    #pragma unroll 1
    for (int jj = 0; jj < TPW; jj++) {
      const int j = wv * TPW + jj;
      const int n = j * 16 + lq;
      v8f acc;
      { const float bv = s_bo1[n];
        #pragma unroll
        for (int r = 0; r < 8; r++) acc[r] = bv; }
      #pragma unroll 4
      for (int k = 0; k < L_SZ; k += 32) {
        v16bf a = load_a_lds(xa, lane, k);
        v16bf b = load_b_g(Wo1, L_SZ, j * 16, lane, k);
        acc = WMMA_BF16(a, b, acc);
      }
      #pragma unroll
      for (int r = 0; r < 8; r++)
        h1[(half * 8 + r) * PITCH + n] = f2bf(tanh_fast(acc[r]));
    }
    __syncthreads();

    // ===== ODE stage 2: x += dt * (h1 @ Wo2^T + bo2); re-stage xa = bf16(x)
    #pragma unroll 1
    for (int jj = 0; jj < TPW; jj++) {
      const int j = wv * TPW + jj;
      const int n = j * 16 + lq;
      v8f acc;
      { const float bv = s_bo2[n];
        #pragma unroll
        for (int r = 0; r < 8; r++) acc[r] = bv; }
      #pragma unroll 4
      for (int k = 0; k < L_SZ; k += 32) {   // K = H_ODE = 512
        v16bf a = load_a_lds(h1, lane, k);
        v16bf b = load_b_g(Wo2, L_SZ, j * 16, lane, k);
        acc = WMMA_BF16(a, b, acc);
      }
      #pragma unroll
      for (int r = 0; r < 8; r++) {
        const int row = half * 8 + r;
        const float xn = xst[jj][r] + s_dt[row] * acc[r];
        xst[jj][r] = xn;
        xa[row * PITCH + n] = f2bf(xn);
      }
    }
    __syncthreads();

    // ===== LSTM cell: g = xi @ W_ih^T + x @ W_hh^T + (b_ih+b_hh), gates i,f,g,o
    #pragma unroll 1
    for (int jj = 0; jj < TPW; jj++) {
      const int j = wv * TPW + jj;
      const int n = j * 16 + lq;
      v8f ai, af, ag, ao;
      { const float bi = s_bsum[n], bff = s_bsum[L_SZ + n],
                    bg = s_bsum[2 * L_SZ + n], bo = s_bsum[3 * L_SZ + n];
        #pragma unroll
        for (int r = 0; r < 8; r++) { ai[r] = bi; af[r] = bff; ag[r] = bg; ao[r] = bo; } }
      #pragma unroll 2
      for (int k = 0; k < 2 * D_SZ; k += 32) {          // xi @ W_ih^T  (K = 2D = 512)
        v16bf a = load_a_lds(xs, lane, k);
        ai = WMMA_BF16(a, load_b_g(Wih, 2 * D_SZ, 0 * L_SZ + j * 16, lane, k), ai);
        af = WMMA_BF16(a, load_b_g(Wih, 2 * D_SZ, 1 * L_SZ + j * 16, lane, k), af);
        ag = WMMA_BF16(a, load_b_g(Wih, 2 * D_SZ, 2 * L_SZ + j * 16, lane, k), ag);
        ao = WMMA_BF16(a, load_b_g(Wih, 2 * D_SZ, 3 * L_SZ + j * 16, lane, k), ao);
      }
      #pragma unroll 2
      for (int k = 0; k < L_SZ; k += 32) {              // x @ W_hh^T   (K = L = 512)
        v16bf a = load_a_lds(xa, lane, k);
        ai = WMMA_BF16(a, load_b_g(Whh, L_SZ, 0 * L_SZ + j * 16, lane, k), ai);
        af = WMMA_BF16(a, load_b_g(Whh, L_SZ, 1 * L_SZ + j * 16, lane, k), af);
        ag = WMMA_BF16(a, load_b_g(Whh, L_SZ, 2 * L_SZ + j * 16, lane, k), ag);
        ao = WMMA_BF16(a, load_b_g(Whh, L_SZ, 3 * L_SZ + j * 16, lane, k), ao);
      }
      #pragma unroll
      for (int r = 0; r < 8; r++) {
        const int row = half * 8 + r;
        if (s_mflag[row]) {                  // update only where mask row has data
          const float c_old = cst[jj][r];
          const float c2 = sigf(af[r]) * c_old + sigf(ai[r]) * tanh_fast(ag[r]);
          cst[jj][r] = c2;
          xst[jj][r] = sigf(ao[r]) * tanh_fast(c2);
        }
      }
    }
    __syncthreads();
  }

  // write final latent state x [16,512] -> global
  #pragma unroll
  for (int jj = 0; jj < TPW; jj++) {
    const int n = (wv * TPW + jj) * 16 + lq;
    #pragma unroll
    for (int r = 0; r < 8; r++)
      xfin[(size_t)(bbase + half * 8 + r) * L_SZ + n] = xst[jj][r];
  }
}

// ---------------------------------------------------------------------------
// small helper kernels
// ---------------------------------------------------------------------------
__global__ void cvt_bf16_kernel(const float* __restrict__ src,
                                unsigned short* __restrict__ dst, int n) {
  const int i = blockIdx.x * blockDim.x + threadIdx.x;
  if (i < n) dst[i] = f2bf(src[i]);
}

// u[b,j] = tanh(x[b,:] . Wt1[j,:] + bt1[j]),  j < 100
__global__ void head1_kernel(const float* __restrict__ xfin, const float* __restrict__ Wt1,
                             const float* __restrict__ bt1, float* __restrict__ u) {
  const int idx = blockIdx.x * blockDim.x + threadIdx.x;
  if (idx >= B_SZ * 100) return;
  const int b = idx / 100, j = idx % 100;
  const float* x = xfin + (size_t)b * L_SZ;
  const float* w = Wt1 + (size_t)j * L_SZ;
  float acc = bt1[j];
  for (int k = 0; k < L_SZ; k++) acc += x[k] * w[k];
  u[idx] = tanh_fast(acc);
}

// z[b,n] = u[b,:] . Wt2[n,:] + bt2[n]; out = [mean_z0 | softplus(std_raw)]
__global__ void head2_kernel(const float* __restrict__ u, const float* __restrict__ Wt2,
                             const float* __restrict__ bt2, float* __restrict__ out) {
  const int idx = blockIdx.x * blockDim.x + threadIdx.x;
  if (idx >= B_SZ * 1024) return;
  const int b = idx >> 10, n = idx & 1023;
  const float* uu = u + (size_t)b * 100;
  const float* w  = Wt2 + (size_t)n * 100;
  float acc = bt2[n];
  for (int k = 0; k < 100; k++) acc += uu[k] * w[k];
  if (n < 512) {
    out[(size_t)b * 512 + n] = acc;                              // mean_z0
  } else {
    const float sp = (acc > 20.0f) ? acc : log1pf(__expf(acc));  // softplus(std)
    out[(size_t)B_SZ * 512 + (size_t)b * 512 + (n - 512)] = sp;
  }
}

// ---------------------------------------------------------------------------
extern "C" void kernel_launch(void* const* d_in, const int* in_sizes, int n_in,
                              void* d_out, int out_size, void* d_ws, size_t ws_size,
                              hipStream_t stream) {
  const float* truth = (const float*)d_in[0];
  const float* maskp = (const float*)d_in[1];
  const float* ts    = (const float*)d_in[2];
  const float* W_ih  = (const float*)d_in[3];
  const float* W_hh  = (const float*)d_in[4];
  const float* b_ih  = (const float*)d_in[5];
  const float* b_hh  = (const float*)d_in[6];
  const float* Wt1   = (const float*)d_in[7];
  const float* bt1   = (const float*)d_in[8];
  const float* Wt2   = (const float*)d_in[9];
  const float* bt2   = (const float*)d_in[10];
  const float* Wo1   = (const float*)d_in[11];
  const float* bo1   = (const float*)d_in[12];
  const float* Wo2   = (const float*)d_in[13];
  const float* bo2   = (const float*)d_in[14];
  float* out = (float*)d_out;

  // workspace layout (bytes)
  char* ws = (char*)d_ws;
  unsigned short* Wih_bf = (unsigned short*)(ws);                    // 2048*512*2 = 2 MiB
  unsigned short* Whh_bf = (unsigned short*)(ws + (2u << 20));       // 2 MiB
  unsigned short* Wo1_bf = (unsigned short*)(ws + (4u << 20));       // 512 KiB
  unsigned short* Wo2_bf = (unsigned short*)(ws + (4u << 20) + (512u << 10));
  float* xfin = (float*)(ws + (5u << 20));                           // 512*512*4 = 1 MiB
  float* ubuf = (float*)(ws + (6u << 20));                           // 512*100*4

  const int NW1 = G_SZ * 2 * D_SZ;   // 2048*512
  const int NW2 = L_SZ * L_SZ;       // 512*512
  cvt_bf16_kernel<<<(NW1 + 255) / 256, 256, 0, stream>>>(W_ih, Wih_bf, NW1);
  cvt_bf16_kernel<<<(NW1 + 255) / 256, 256, 0, stream>>>(W_hh, Whh_bf, NW1);
  cvt_bf16_kernel<<<(NW2 + 255) / 256, 256, 0, stream>>>(Wo1, Wo1_bf, NW2);
  cvt_bf16_kernel<<<(NW2 + 255) / 256, 256, 0, stream>>>(Wo2, Wo2_bf, NW2);

  ode_rnn_scan_kernel<<<B_SZ / 16, THREADS, 0, stream>>>(
      truth, maskp, ts, Wih_bf, Whh_bf, Wo1_bf, Wo2_bf,
      b_ih, b_hh, bo1, bo2, xfin);

  head1_kernel<<<(B_SZ * 100 + 255) / 256, 256, 0, stream>>>(xfin, Wt1, bt1, ubuf);
  head2_kernel<<<(B_SZ * 1024 + 255) / 256, 256, 0, stream>>>(ubuf, Wt2, bt2, out);
}
